// MemoryEfficientAttention_20366734917758
// MI455X (gfx1250) — matrix operand order repaired and checked by reading, not compiled, not run on hardware
//
#include <hip/hip_runtime.h>
#include <cstdint>
#include <cstddef>

// ---------- CDNA5 WMMA types ----------
typedef __attribute__((ext_vector_type(16))) __bf16 v16bf;
typedef __attribute__((ext_vector_type(8)))  float  v8f;

union Frag {           // 16x32 (A) or 32x16 (B) bf16 fragment: 8 VGPRs
  v16bf bf;
  uint4 q[2];
};

// Native RNE f32->bf16 (lowers to hardware cvt; pairs may fuse to v_cvt_pk)
__device__ __forceinline__ unsigned short f2bf(float f) {
  return __builtin_bit_cast(unsigned short, (__bf16)f);
}

__device__ __forceinline__ float bf2f(unsigned short b) {
  union { unsigned int u; float f; } v; v.u = ((unsigned int)b) << 16;
  return v.f;
}

// Hardware exp2: v_exp_f32 (no range-reduction multiply)
__device__ __forceinline__ float fast_exp2(float x) {
  return __builtin_amdgcn_exp2f(x);
}

// A-matrix (16x32, MxK) fragment, row-major source, lda in elements.
// ISA layout: lanes 0-15 M=lane hold K=0..7,16..23; lanes 16-31 hold K=8..15,24..31.
__device__ __forceinline__ Frag load_a(const unsigned short* base, int lda) {
  const int lane = threadIdx.x & 31;
  const int hi = lane >> 4;
  const unsigned short* row = base + (size_t)(lane & 15) * lda;
  Frag f;
  f.q[0] = *(const uint4*)(row + hi * 8);
  f.q[1] = *(const uint4*)(row + 16 + hi * 8);
  return f;
}

// B-matrix (32x16, KxN) fragment where B[k,n] = Wrow[n][k] (W row-major, ldw elems).
// Lanes 0-15 (N=lane) hold K=0..7,16..23; lanes 16-31 hold K=8..15,24..31.
__device__ __forceinline__ Frag load_b(const unsigned short* base, int ldw) {
  const int lane = threadIdx.x & 31;
  const int hi = lane >> 4;
  const unsigned short* row = base + (size_t)(lane & 15) * ldw;
  Frag f;
  f.q[0] = *(const uint4*)(row + hi * 8);
  f.q[1] = *(const uint4*)(row + 16 + hi * 8);
  return f;
}

__device__ __forceinline__ v8f wmma_bf16(const Frag& a, const Frag& b, v8f c) {
  return __builtin_amdgcn_wmma_f32_16x16x32_bf16(false, a.bf, false, b.bf,
                                                 (short)0, c, false, false);
}

// ---------- fp32 -> bf16 convert ----------
__global__ void cvt_f32_bf16(const float* __restrict__ in,
                             unsigned short* __restrict__ out, int n) {
  int i = blockIdx.x * blockDim.x + threadIdx.x;
  if (i < n) out[i] = f2bf(in[i]);
}

// ---------- generic bf16 WMMA GEMM: C = A @ W^T (+bias) ----------
// A: [M,K] bf16 row-major, W: [N,K] bf16 row-major.
// FINAL=false -> bf16 out; FINAL=true -> fp32 out + bias.
// Block: 128 threads = 4 waves (2x2), block tile 128x128, wave tile 64x64.
template <bool FINAL>
__global__ __launch_bounds__(128) void gemm_bf16_wmma(
    const unsigned short* __restrict__ A, const unsigned short* __restrict__ W,
    unsigned short* __restrict__ Cb, float* __restrict__ Cf,
    const float* __restrict__ bias, int M, int N, int K) {
  const int wave = threadIdx.x >> 5;
  const int lane = threadIdx.x & 31;
  const int m0 = blockIdx.y * 128 + (wave >> 1) * 64;
  const int n0 = blockIdx.x * 128 + (wave & 1) * 64;

  v8f acc[4][4];
#pragma unroll
  for (int mi = 0; mi < 4; ++mi)
#pragma unroll
    for (int ni = 0; ni < 4; ++ni)
      acc[mi][ni] = (v8f){0.f, 0.f, 0.f, 0.f, 0.f, 0.f, 0.f, 0.f};

  for (int k0 = 0; k0 < K; k0 += 32) {
    Frag a[4], b[4];
#pragma unroll
    for (int mi = 0; mi < 4; ++mi)
      a[mi] = load_a(A + (size_t)(m0 + mi * 16) * K + k0, K);
#pragma unroll
    for (int ni = 0; ni < 4; ++ni)
      b[ni] = load_b(W + (size_t)(n0 + ni * 16) * K + k0, K);
#pragma unroll
    for (int mi = 0; mi < 4; ++mi)
#pragma unroll
      for (int ni = 0; ni < 4; ++ni)
        acc[mi][ni] = wmma_bf16(a[mi], b[ni], acc[mi][ni]);
  }

  const int hi = lane >> 4, col = lane & 15;
#pragma unroll
  for (int mi = 0; mi < 4; ++mi)
#pragma unroll
    for (int ni = 0; ni < 4; ++ni) {
      const int n = n0 + ni * 16 + col;
      float bv = FINAL ? bias[n] : 0.f;
#pragma unroll
      for (int r = 0; r < 8; ++r) {
        const int m = m0 + mi * 16 + r + hi * 8;
        float v = acc[mi][ni][r] + bv;
        if (FINAL) Cf[(size_t)m * N + n] = v;
        else       Cb[(size_t)m * N + n] = f2bf(v);
      }
    }
}

// ---------- split qkv [B,T,3C] -> Q[b,h,t,d] (pre-scaled), K[b,h,t,d], Vt[b,h,d,t]
// Q is scaled by (1/sqrt(D)) * log2(e) so attention runs softmax in exp2 domain
// with no per-score multiplies.
__global__ void pack_qkv(const unsigned short* __restrict__ qkv,
                         unsigned short* __restrict__ Q,
                         unsigned short* __restrict__ Kh,
                         unsigned short* __restrict__ Vt) {
  const int T = 2048, C = 1024, H = 16, D = 64;
  const float qscale = 0.125f * 1.4426950408889634f;  // 1/sqrt(64) * log2(e)
  size_t idx = (size_t)blockIdx.x * blockDim.x + threadIdx.x;
  const size_t total = (size_t)2 * T * 3 * C;
  if (idx >= total) return;
  int c3 = (int)(idx % (3 * C));
  size_t bt = idx / (3 * C);
  int t = (int)(bt % T);
  int b = (int)(bt / T);
  int which = c3 / C, c = c3 % C;
  int h = c / D, d = c % D;
  unsigned short v = qkv[idx];
  int bh = b * H + h;
  if (which == 0) {
    Q[((size_t)bh * T + t) * D + d] = f2bf(bf2f(v) * qscale);
  } else if (which == 1) {
    Kh[((size_t)bh * T + t) * D + d] = v;
  } else {
    Vt[((size_t)bh * D + d) * T + t] = v;
  }
}

// ---------- flash attention: one wave = 16 query rows of one (b,h) ----------
__global__ __launch_bounds__(256) void flash_attn(
    const unsigned short* __restrict__ Qh, const unsigned short* __restrict__ Kh,
    const unsigned short* __restrict__ Vt, unsigned short* __restrict__ Ob) {
  const int T = 2048, D = 64, H = 16, C = 1024;
  __shared__ __align__(16) unsigned short lds[8 * 16 * 32];  // 1KB per wave
  const int wave = threadIdx.x >> 5;
  const int lane = threadIdx.x & 31;
  unsigned short* P = &lds[wave * 512];

  const int gw = blockIdx.x * 8 + wave;
  const int bh = gw >> 7;           // T/16 = 128 query tiles per (b,h)
  const int m0 = (gw & 127) * 16;

  const unsigned short* Q  = Qh + (size_t)bh * T * D;
  const unsigned short* Kp = Kh + (size_t)bh * T * D;
  const unsigned short* Vp = Vt + (size_t)bh * D * T;

  const Frag qf0 = load_a(Q + (size_t)m0 * D, D);       // d = 0..31 (pre-scaled)
  const Frag qf1 = load_a(Q + (size_t)m0 * D + 32, D);  // d = 32..63

  // B-matrix of bf16 ones: row-sum of P via one WMMA (replaces shuffle reduce).
  // Pin in VGPRs (component-wise tied 32-bit operands) so it isn't re-splatted
  // every loop iteration.
  unsigned int o0 = 0x3F803F80u, o1 = 0x3F803F80u, o2 = 0x3F803F80u,
               o3 = 0x3F803F80u, o4 = 0x3F803F80u, o5 = 0x3F803F80u,
               o6 = 0x3F803F80u, o7 = 0x3F803F80u;
  asm volatile("" : "+v"(o0), "+v"(o1), "+v"(o2), "+v"(o3),
                    "+v"(o4), "+v"(o5), "+v"(o6), "+v"(o7));
  Frag onesb;
  onesb.q[0] = make_uint4(o0, o1, o2, o3);
  onesb.q[1] = make_uint4(o4, o5, o6, o7);

  v8f o[4];
#pragma unroll
  for (int di = 0; di < 4; ++di)
    o[di] = (v8f){0.f, 0.f, 0.f, 0.f, 0.f, 0.f, 0.f, 0.f};
  float mrow[8], lrow[8];
#pragma unroll
  for (int r = 0; r < 8; ++r) { mrow[r] = -3.0e38f; lrow[r] = 0.f; }

  const int hi = lane >> 4, col = lane & 15;
  const v8f z8 = (v8f){0.f, 0.f, 0.f, 0.f, 0.f, 0.f, 0.f, 0.f};

#pragma unroll 2
  for (int kt = 0; kt < T; kt += 32) {
    // scores (log2 domain): S[16x32] = Qscaled(16x64) @ K(kt..kt+31, 64)^T
    Frag kb00 = load_b(Kp + (size_t)kt * D, D);
    Frag kb01 = load_b(Kp + (size_t)kt * D + 32, D);
    Frag kb10 = load_b(Kp + (size_t)(kt + 16) * D, D);
    Frag kb11 = load_b(Kp + (size_t)(kt + 16) * D + 32, D);
    v8f s0 = wmma_bf16(qf0, kb00, z8); s0 = wmma_bf16(qf1, kb01, s0);
    v8f s1 = wmma_bf16(qf0, kb10, z8); s1 = wmma_bf16(qf1, kb11, s1);

    float alpha[8];
#pragma unroll
    for (int r = 0; r < 8; ++r) {
      float mx = fmaxf(s0[r], s1[r]);                 // row max across 16 lanes
      mx = fmaxf(mx, __shfl_xor(mx, 1, 32));
      mx = fmaxf(mx, __shfl_xor(mx, 2, 32));
      mx = fmaxf(mx, __shfl_xor(mx, 4, 32));
      mx = fmaxf(mx, __shfl_xor(mx, 8, 32));
      float mnew = fmaxf(mrow[r], mx);
      alpha[r] = fast_exp2(mrow[r] - mnew);
      mrow[r] = mnew;
      s0[r] = fast_exp2(s0[r] - mnew);
      s1[r] = fast_exp2(s1[r] - mnew);
    }
#pragma unroll
    for (int di = 0; di < 4; ++di)
#pragma unroll
      for (int r = 0; r < 8; ++r) o[di][r] *= alpha[r];

    // C-layout P tile -> LDS (row-major 16x32) -> reload in A-layout.
    asm volatile("s_wait_dscnt 0" ::: "memory");   // prior A-layout reads done
#pragma unroll
    for (int r = 0; r < 8; ++r) {
      const int row = r + hi * 8;
      P[row * 32 + col]      = f2bf(s0[r]);
      P[row * 32 + 16 + col] = f2bf(s1[r]);
    }
    asm volatile("s_wait_dscnt 0" ::: "memory");   // stores visible before loads
    Frag pf = load_a(P, 32);

    // row-sum of P via WMMA with ones: every column of ssum = rowsum
    v8f ssum = wmma_bf16(pf, onesb, z8);
#pragma unroll
    for (int r = 0; r < 8; ++r) lrow[r] = lrow[r] * alpha[r] + ssum[r];

    // O(16x64) += P(16x32) @ V(kt..kt+31, 0..63); Vt rows are d, contiguous in t
#pragma unroll
    for (int di = 0; di < 4; ++di) {
      Frag vb = load_b(Vp + (size_t)(di * 16) * T + kt, T);
      o[di] = wmma_bf16(pf, vb, o[di]);
    }
  }

  const int b = bh >> 4, h = bh & 15;
#pragma unroll
  for (int di = 0; di < 4; ++di)
#pragma unroll
    for (int r = 0; r < 8; ++r) {
      const int t = m0 + r + hi * 8;
      float v = o[di][r] / lrow[r];
      Ob[((size_t)(b * T + t)) * C + h * 64 + di * 16 + col] = f2bf(v);
    }
}

// ---------- host launcher ----------
extern "C" void kernel_launch(void* const* d_in, const int* in_sizes, int n_in,
                              void* d_out, int out_size, void* d_ws, size_t ws_size,
                              hipStream_t stream) {
  const float* x     = (const float*)d_in[0];
  const float* w_qkv = (const float*)d_in[1];
  const float* w_out = (const float*)d_in[2];
  const float* b_out = (const float*)d_in[3];
  float* out = (float*)d_out;

  const int B = 2, T = 2048, C = 1024, H = 16;
  const int M = B * T;  // 4096

  unsigned short* xb    = (unsigned short*)d_ws;
  unsigned short* wqkvb = xb    + (size_t)M * C;       // 4.19M elems
  unsigned short* woutb = wqkvb + (size_t)3 * C * C;   // 3.15M
  unsigned short* qkvb  = woutb + (size_t)C * C;       // 1.05M
  unsigned short* Qh    = qkvb  + (size_t)M * 3 * C;   // 12.6M
  unsigned short* Kh    = Qh    + (size_t)M * C;
  unsigned short* Vt    = Kh    + (size_t)M * C;
  unsigned short* attnb = Vt    + (size_t)M * C;       // total ~75.5 MB

  cvt_f32_bf16<<<(M * C + 255) / 256, 256, 0, stream>>>(x, xb, M * C);
  cvt_f32_bf16<<<(3 * C * C + 255) / 256, 256, 0, stream>>>(w_qkv, wqkvb, 3 * C * C);
  cvt_f32_bf16<<<(C * C + 255) / 256, 256, 0, stream>>>(w_out, woutb, C * C);

  // qkv = x @ w_qkv^T  -> bf16
  gemm_bf16_wmma<false><<<dim3(3 * C / 128, M / 128), 128, 0, stream>>>(
      xb, wqkvb, qkvb, nullptr, nullptr, M, 3 * C, C);

  pack_qkv<<<(int)(((size_t)M * 3 * C + 255) / 256), 256, 0, stream>>>(
      qkvb, Qh, Kh, Vt);

  // B*H*(T/16) = 4096 waves, 8 waves/block -> 512 blocks
  flash_attn<<<(B * H * (T / 16)) / 8, 256, 0, stream>>>(Qh, Kh, Vt, attnb);

  // out = attn @ w_out^T + b_out -> fp32
  gemm_bf16_wmma<true><<<dim3(C / 128, M / 128), 128, 0, stream>>>(
      attnb, woutb, nullptr, out, b_out, M, C, C);
}
